// GradientBoostingDeicisionTreeLayer_49426483642836
// MI455X (gfx1250) — matrix operand order repaired and checked by reading, not compiled, not run on hardware
//
#include <hip/hip_runtime.h>
#include <hip/hip_bf16.h>

typedef __attribute__((ext_vector_type(16))) _Float16 v16h;
typedef __attribute__((ext_vector_type(8)))  float    v8f;

#define T_TREES  200
#define D_FEAT   512
#define N_NODE   64
#define BATCH    2048
#define N_LEAFN  15
#define N_LEAVES 16
#define LR_BOOST 0.01f

// LDS strides (halves) keeping 16B-aligned b128 access.
#define XS_STR 40   // 32 halves + pad (80B rows)
#define WS_STR 40
#define HS_STR 65
#define LT_STR 80   // transpose kernel tile stride (160B rows)

// ---- CDNA5 async global->LDS copy (ASYNCcnt-tracked, no VGPR data path) ----
#if defined(__has_builtin)
#if __has_builtin(__builtin_amdgcn_global_load_async_to_lds_b128)
#define HAVE_ASYNC_LDS 1
#endif
#endif

#ifdef HAVE_ASYNC_LDS
typedef int v4i_vs __attribute__((vector_size(16)));
typedef __attribute__((address_space(1))) v4i_vs* glb_v4p;
typedef __attribute__((address_space(3))) v4i_vs* lds_v4p;

__device__ __forceinline__ void async_copy_b128(const void* g, void* l) {
  __builtin_amdgcn_global_load_async_to_lds_b128(
      (glb_v4p)(uintptr_t)g,
      (lds_v4p)(unsigned)(uintptr_t)l,
      /*imm offset=*/0, /*cpol=*/0);
}
__device__ __forceinline__ void wait_async_zero() {
#if __has_builtin(__builtin_amdgcn_s_wait_asynccnt)
  __builtin_amdgcn_s_wait_asynccnt(0);
#else
  asm volatile("s_wait_asynccnt 0x0" ::: "memory");
#endif
}
#endif

// -------- kernel 1a: fp32 -> fp16 convert of x (elementwise, vectorized) -----
__global__ __launch_bounds__(256) void gbdt_cvt_x(
    const float* __restrict__ x, _Float16* __restrict__ xb, int n4)
{
  int i = blockIdx.x * blockDim.x + threadIdx.x;
  if (i >= n4) return;
  float4 v = reinterpret_cast<const float4*>(x)[i];
  union { uint2 u; _Float16 h[4]; } o;
  o.h[0] = (_Float16)v.x; o.h[1] = (_Float16)v.y;
  o.h[2] = (_Float16)v.z; o.h[3] = (_Float16)v.w;
  reinterpret_cast<uint2*>(xb)[i] = o.u;
}

// ---- kernel 1b: fp32 [T][D][N] -> fp16 [T][N][D] tiled transpose-convert ----
__global__ __launch_bounds__(256) void gbdt_cvt_w_tr(
    const float* __restrict__ wt, _Float16* __restrict__ wb)
{
  __shared__ _Float16 lt[64 * LT_STR];
  const int t  = blockIdx.x;
  const int d0 = blockIdx.y * 64;
  const int i  = threadIdx.x;

  {
    int dr   = i >> 2;           // 0..63
    int nseg = (i & 3) * 16;     // 0,16,32,48
    const float* src = wt + ((size_t)t * D_FEAT + d0 + dr) * N_NODE + nseg;
    union { uint4 q[2]; _Float16 h[16]; } o;
#pragma unroll
    for (int j = 0; j < 16; ++j) o.h[j] = (_Float16)src[j];
    *reinterpret_cast<uint4*>(&lt[dr * LT_STR + nseg])     = o.q[0];
    *reinterpret_cast<uint4*>(&lt[dr * LT_STR + nseg + 8]) = o.q[1];
  }
  __syncthreads();
  {
    int n    = i >> 2;           // 0..63
    int dseg = (i & 3) * 16;     // 0,16,32,48
    union { uint4 q[2]; _Float16 h[16]; } o;
#pragma unroll
    for (int j = 0; j < 16; ++j) o.h[j] = lt[(dseg + j) * LT_STR + n];
    _Float16* dst = wb + ((size_t)t * N_NODE + n) * D_FEAT + d0 + dseg;
    *reinterpret_cast<uint4*>(dst)     = o.q[0];
    *reinterpret_cast<uint4*>(dst + 8) = o.q[1];
  }
}

// ------------- kernel 2: fused per-tree forward (WMMA GEMM + tree) -----------
// grid = (BATCH/32, T_TREES), block = 128 (4 wave32). Wave w owns node slice
// [16w, 16w+16) and two 16-row batch tiles. LDS ping-pong; tile k+1 is copied
// (async-to-LDS when available) while tile k feeds the WMMAs.
__global__ __launch_bounds__(128) void gbdt_tree_fwd(
    const _Float16* __restrict__ xb,   // [BATCH][D_FEAT]
    const _Float16* __restrict__ wb,   // [T][N_NODE][D_FEAT]  (node-major!)
    const float* __restrict__ bt,      // [T][N_NODE]
    const float* __restrict__ wd,      // [T][N_NODE][15]
    const float* __restrict__ bd,      // [T][15]
    const float* __restrict__ wl,      // [T][16]
    const float* __restrict__ bl,      // [T]
    float* __restrict__ fbuf)          // [BATCH][T]
{
  __shared__ _Float16 xs[2][32 * XS_STR];
  __shared__ _Float16 wsT[2][N_NODE * WS_STR];
  __shared__ float    hs[32 * HS_STR];
  __shared__ float    ps[32 * 16];

  const int tid  = threadIdx.x;
  const int lane = tid & 31;
  const int wv   = tid >> 5;
  const int t    = blockIdx.y;
  const int m0   = blockIdx.x * 32;
  const int n0   = wv * 16;

  // warm L2 for the post-GEMM decision layer weights of this tree
  __builtin_prefetch(wd + (size_t)t * N_NODE * N_LEAFN + tid, 0, 1);

  // cooperative-copy mappings (16B chunks)
  const int xrow = tid >> 2;               // 0..31
  const int xkk  = (tid & 3) * 8;          // 0,8,16,24
  const _Float16* xsrc = xb + (size_t)(m0 + xrow) * D_FEAT + xkk;
  const int wn   = tid >> 1;               // 0..63
  const int wkk  = (tid & 1) * 16;         // 0,16
  const _Float16* wsrc = wb + ((size_t)t * N_NODE + wn) * D_FEAT + wkk;

  _Float16* xdst0 = &xs[0][xrow * XS_STR + xkk];
  _Float16* xdst1 = &xs[1][xrow * XS_STR + xkk];
  _Float16* wdst0 = &wsT[0][wn * WS_STR + wkk];
  _Float16* wdst1 = &wsT[1][wn * WS_STR + wkk];

  // fragment addressing (ISA 16-bit A 16x32 / B 32x16 wave32 layouts)
  const int arow  = lane & 15;
  const int akoff = (lane < 16) ? 0 : 8;
  const int bcol  = n0 + (lane & 15);
  const int bkoff = (lane < 16) ? 0 : 8;

  v8f acc0 = {};
  v8f acc1 = {};

#ifdef HAVE_ASYNC_LDS
  // ---------------- async-to-LDS pipeline ----------------
  async_copy_b128(xsrc, xdst0);
  async_copy_b128(wsrc, wdst0);
  async_copy_b128(wsrc + 8, wdst0 + 8);
  wait_async_zero();
  __syncthreads();

#pragma unroll
  for (int k = 0; k < D_FEAT / 32; ++k) {
    const int cur = k & 1;
    if (k < D_FEAT / 32 - 1) {          // issue copies for tile k+1
      const int koff = (k + 1) * 32;
      _Float16* xd = cur ? xdst0 : xdst1;
      _Float16* wd2 = cur ? wdst0 : wdst1;
      async_copy_b128(xsrc + koff, xd);
      async_copy_b128(wsrc + koff, wd2);
      async_copy_b128(wsrc + koff + 8, wd2 + 8);
    }

    union { v16h v; uint4 q[2]; } a0, a1, b;
    a0.q[0] = *reinterpret_cast<const uint4*>(&xs[cur][arow * XS_STR + akoff]);
    a0.q[1] = *reinterpret_cast<const uint4*>(&xs[cur][arow * XS_STR + akoff + 16]);
    a1.q[0] = *reinterpret_cast<const uint4*>(&xs[cur][(16 + arow) * XS_STR + akoff]);
    a1.q[1] = *reinterpret_cast<const uint4*>(&xs[cur][(16 + arow) * XS_STR + akoff + 16]);
    b.q[0]  = *reinterpret_cast<const uint4*>(&wsT[cur][bcol * WS_STR + bkoff]);
    b.q[1]  = *reinterpret_cast<const uint4*>(&wsT[cur][bcol * WS_STR + bkoff + 16]);

    acc0 = __builtin_amdgcn_wmma_f32_16x16x32_f16(false, a0.v, false, b.v,
                                                  (short)0, acc0, false, false);
    acc1 = __builtin_amdgcn_wmma_f32_16x16x32_f16(false, a1.v, false, b.v,
                                                  (short)0, acc1, false, false);

    if (k < D_FEAT / 32 - 1) wait_async_zero();
    __syncthreads();
  }
#else
  // ---------------- register-staged fallback pipeline ----------------
  uint4 rx  = *reinterpret_cast<const uint4*>(xsrc);
  uint4 rw0 = *reinterpret_cast<const uint4*>(wsrc);
  uint4 rw1 = *reinterpret_cast<const uint4*>(wsrc + 8);
  *reinterpret_cast<uint4*>(xdst0)     = rx;
  *reinterpret_cast<uint4*>(wdst0)     = rw0;
  *reinterpret_cast<uint4*>(wdst0 + 8) = rw1;
  __syncthreads();

#pragma unroll
  for (int k = 0; k < D_FEAT / 32; ++k) {
    const int cur = k & 1;
    if (k < D_FEAT / 32 - 1) {
      const int koff = (k + 1) * 32;
      rx  = *reinterpret_cast<const uint4*>(xsrc + koff);
      rw0 = *reinterpret_cast<const uint4*>(wsrc + koff);
      rw1 = *reinterpret_cast<const uint4*>(wsrc + koff + 8);
    }

    union { v16h v; uint4 q[2]; } a0, a1, b;
    a0.q[0] = *reinterpret_cast<const uint4*>(&xs[cur][arow * XS_STR + akoff]);
    a0.q[1] = *reinterpret_cast<const uint4*>(&xs[cur][arow * XS_STR + akoff + 16]);
    a1.q[0] = *reinterpret_cast<const uint4*>(&xs[cur][(16 + arow) * XS_STR + akoff]);
    a1.q[1] = *reinterpret_cast<const uint4*>(&xs[cur][(16 + arow) * XS_STR + akoff + 16]);
    b.q[0]  = *reinterpret_cast<const uint4*>(&wsT[cur][bcol * WS_STR + bkoff]);
    b.q[1]  = *reinterpret_cast<const uint4*>(&wsT[cur][bcol * WS_STR + bkoff + 16]);

    acc0 = __builtin_amdgcn_wmma_f32_16x16x32_f16(false, a0.v, false, b.v,
                                                  (short)0, acc0, false, false);
    acc1 = __builtin_amdgcn_wmma_f32_16x16x32_f16(false, a1.v, false, b.v,
                                                  (short)0, acc1, false, false);

    if (k < D_FEAT / 32 - 1) {
      _Float16* xd = cur ? xdst0 : xdst1;
      _Float16* wd2 = cur ? wdst0 : wdst1;
      *reinterpret_cast<uint4*>(xd)      = rx;
      *reinterpret_cast<uint4*>(wd2)     = rw0;
      *reinterpret_cast<uint4*>(wd2 + 8) = rw1;
    }
    __syncthreads();
  }
#endif

  // ---- bias + ReLU, scatter both C tiles to LDS ----
  {
    const int col   = n0 + (lane & 15);
    const int rbase = (lane < 16) ? 0 : 8;
    const float bias = bt[t * N_NODE + col];
#pragma unroll
    for (int r = 0; r < 8; ++r) {
      hs[(rbase + r) * HS_STR + col]      = fmaxf(acc0[r] + bias, 0.0f);
      hs[(16 + rbase + r) * HS_STR + col] = fmaxf(acc1[r] + bias, 0.0f);
    }
  }
  __syncthreads();

  // ---- decision layer: p = sigmoid(h @ w_d + b_d), 32 rows x 15 nodes ----
  {
    const int row = tid & 31;
    for (int l = tid >> 5; l < N_LEAFN; l += 4) {
      float s = bd[t * N_LEAFN + l];
#pragma unroll 8
      for (int n = 0; n < N_NODE; ++n)
        s += hs[row * HS_STR + n] * wd[((size_t)t * N_NODE + n) * N_LEAFN + l];
      ps[row * 16 + l] = 1.0f / (1.0f + __expf(-s));
    }
  }
  __syncthreads();

  // ---- soft routing over 16 leaves + tanh leaf score ----
  if (tid < 32) {
    const int row = tid;
    float fsum = bl[t];
#pragma unroll
    for (int leaf = 0; leaf < N_LEAVES; ++leaf) {
      float mu = 1.0f;
      int node = 0;
#pragma unroll
      for (int d = 0; d < 4; ++d) {
        int bit = (leaf >> (3 - d)) & 1;
        float pv = ps[row * 16 + node];
        mu *= bit ? (1.0f - pv) : pv;
        node = 2 * node + 1 + bit;
      }
      fsum += mu * wl[t * N_LEAVES + leaf];
    }
    fbuf[(size_t)(m0 + row) * T_TREES + t] = tanhf(fsum);
  }
}

// ---------------- kernel 3: sequential boosting accumulation -----------------
__global__ __launch_bounds__(256) void gbdt_boost_cumsum(
    const float* __restrict__ fbuf,
    const float* __restrict__ f0,
    float* __restrict__ out)
{
  int b = blockIdx.x * blockDim.x + threadIdx.x;
  if (b >= BATCH) return;
  float acc = f0[0];
  out[(size_t)b * (T_TREES + 1)] = acc;
  const float* fr = fbuf + (size_t)b * T_TREES;
  for (int t = 0; t < T_TREES; ++t) {
    acc += LR_BOOST * fr[t];
    out[(size_t)b * (T_TREES + 1) + t + 1] = acc;
  }
}

extern "C" void kernel_launch(void* const* d_in, const int* in_sizes, int n_in,
                              void* d_out, int out_size, void* d_ws, size_t ws_size,
                              hipStream_t stream) {
  const float* x   = (const float*)d_in[0];
  // d_in[1] = y_true (unused in forward)
  const float* w_t = (const float*)d_in[2];
  const float* b_t = (const float*)d_in[3];
  const float* w_d = (const float*)d_in[4];
  const float* b_d = (const float*)d_in[5];
  const float* w_l = (const float*)d_in[6];
  const float* b_l = (const float*)d_in[7];
  const float* f_0 = (const float*)d_in[8];
  float* out = (float*)d_out;

  const int nx = BATCH * D_FEAT;
  const int nw = T_TREES * D_FEAT * N_NODE;

  char* ws = (char*)d_ws;
  _Float16* xb = (_Float16*)ws;
  _Float16* wb = xb + nx;
  float* fbuf  = (float*)(wb + nw);

  gbdt_cvt_x<<<(nx / 4 + 255) / 256, 256, 0, stream>>>(x, xb, nx / 4);
  {
    dim3 grid(T_TREES, D_FEAT / 64);
    gbdt_cvt_w_tr<<<grid, 256, 0, stream>>>(w_t, wb);
  }
  {
    dim3 grid(BATCH / 32, T_TREES);
    gbdt_tree_fwd<<<grid, 128, 0, stream>>>(xb, wb, b_t, w_d, b_d, w_l, b_l, fbuf);
  }
  gbdt_boost_cumsum<<<(BATCH + 255) / 256, 256, 0, stream>>>(fbuf, f_0, out);
}
